// TreeLSTMEncoder_59450937311583
// MI455X (gfx1250) — compile-verified, matrix operand order checked by reading, not run on hardware
//
#include <hip/hip_runtime.h>
#include <hip/hip_bf16.h>

// ---------------------------------------------------------------------------
// LSTM over T=512 steps, B=128, D=H=1024, gate order i,f,o,u.
//  - One-time prep: WcatT[4096][2048] bf16, row j4 = g*1024+j holds
//      k<1024  -> Wh[g][k][j]     (recurrent part, A = h)
//      k>=1024 -> Wx[g][k-1024][j] (input part,   A = x_t)
//  - 512 step kernels: gates = [h ; x_t](128x2048) @ WcatT^T -> 128x4096 via
//    v_wmma_f32_16x16x32_bf16, fused sigmoid/tanh + c/h update.
//  - B fragments double-buffered in LDS (one barrier per K-slice, global
//    load of slice i+1 overlapped with WMMAs of slice i).
//  - h ping-pongs between two bf16 buffers; c updated in place.
// ---------------------------------------------------------------------------

typedef __attribute__((ext_vector_type(16))) __bf16 v16bf;
typedef __attribute__((ext_vector_type(8)))  __bf16 v8bf;
typedef __attribute__((ext_vector_type(8)))  float  v8f;
typedef __attribute__((ext_vector_type(4)))  float  v4f;

#define BATCH 128
#define TSEQ  512
#define DIM   1024
#define KTOT  2048      // [h(0..1023) ; x(1024..2047)]
#define GCOLS 4096      // 4 gates * 1024 output columns

__device__ __forceinline__ float fast_sigmoid(float x) {
  return 1.0f / (1.0f + __expf(-x));
}
__device__ __forceinline__ float fast_tanh(float x) {
  return 2.0f / (1.0f + __expf(-2.0f * x)) - 1.0f;
}

__device__ __forceinline__ v16bf cat8(v8bf lo, v8bf hi) {
  return __builtin_shufflevector(lo, hi,
      0,1,2,3,4,5,6,7,8,9,10,11,12,13,14,15);
}

// 4 gate WMMAs for one K-slice, B fragments read from LDS.
__device__ __forceinline__ void gate_mma(const v8bf* lb, v16bf a,
                                         v8f& a0, v8f& a1, v8f& a2, v8f& a3) {
  v16bf b0 = cat8(lb[0],   lb[1]);
  v16bf b1 = cat8(lb[64],  lb[65]);
  v16bf b2 = cat8(lb[128], lb[129]);
  v16bf b3 = cat8(lb[192], lb[193]);
  a0 = __builtin_amdgcn_wmma_f32_16x16x32_bf16(false, a, false, b0, (short)0, a0, false, false);
  a1 = __builtin_amdgcn_wmma_f32_16x16x32_bf16(false, a, false, b1, (short)0, a1, false, false);
  a2 = __builtin_amdgcn_wmma_f32_16x16x32_bf16(false, a, false, b2, (short)0, a2, false, false);
  a3 = __builtin_amdgcn_wmma_f32_16x16x32_bf16(false, a, false, b3, (short)0, a3, false, false);
}

// Build WcatT (bf16). One thread per element.
__global__ __launch_bounds__(256) void prep_weights(const float* __restrict__ Wx,
                                                    const float* __restrict__ Wh,
                                                    __bf16* __restrict__ WcatT) {
  unsigned idx = blockIdx.x * 256u + threadIdx.x;  // 0 .. 4096*2048-1
  unsigned j4 = idx >> 11;                         // output column 0..4095
  unsigned k  = idx & 2047u;                       // K index 0..2047
  unsigned g  = j4 >> 10;
  unsigned j  = j4 & 1023u;
  float w = (k < 1024u) ? Wh[(g << 20) + (k << 10) + j]
                        : Wx[(g << 20) + ((k - 1024u) << 10) + j];
  WcatT[idx] = (__bf16)w;
}

// Zero h0 (bf16) and c (f32). Workspace is poisoned by the harness.
__global__ __launch_bounds__(256) void init_state(__bf16* __restrict__ h0,
                                                  float* __restrict__ c) {
  unsigned idx = blockIdx.x * 256u + threadIdx.x;  // 0 .. 131071
  h0[idx] = (__bf16)0.0f;
  c[idx]  = 0.0f;
}

// One recurrent step. Grid: 64 blocks (one per 16-wide column tile of H),
// 256 threads = 8 waves (one per 16-row tile of B=128).
__global__ __launch_bounds__(256) void lstm_step(
    const __bf16* __restrict__ WcatT,   // [4096][2048]
    const float*  __restrict__ x,       // [B][T][D] f32
    const float*  __restrict__ bias,    // [4][1024]
    const __bf16* __restrict__ h_in,    // [128][1024]
    __bf16*       __restrict__ h_out,   // [128][1024]
    float*        __restrict__ c_state, // [128][1024]
    float*        __restrict__ h_f32,   // [128][1024] (d_out)
    int t, int write_out)
{
  // Double-buffered B fragments in WMMA lane layout: [buf][gate][lane][chunk]
  __shared__ __align__(16) __bf16 ldsB[2][4 * 32 * 16];   // 2 x 4 KB

  const int tid   = (int)threadIdx.x;
  const int lane  = tid & 31;
  const int wave  = tid >> 5;           // row tile 0..7
  const int cbase = (int)blockIdx.x << 4;
  const int half  = lane >> 4;          // fragment lane group
  const int l15   = lane & 15;

  // ---- B staging role: each thread owns one 16B chunk of one gate fragment.
  const int sg     = tid >> 6;          // gate 0..3
  const int sr     = tid & 63;
  const int slane  = sr >> 1;           // fragment lane 0..31
  const int schunk = sr & 1;
  const __bf16* bsrc = WcatT
      + (size_t)((sg << 10) + cbase + (slane & 15)) * KTOT
      + (schunk << 4) + ((slane >> 4) << 3);
  const int sslot = (sg << 6) + (slane << 1) + schunk;   // v8bf slot

  // ---- A sources: h (bf16) for k<1024, x_t (f32 -> bf16) for k>=1024.
  const int arow = (wave << 4) + l15;                    // batch row
  const __bf16* hrow = h_in + arow * DIM + (half << 3);
  const float*  xrow = x + ((size_t)(arow * TSEQ + t) << 10) + (half << 3);

  v8f acc0 = {}, acc1 = {}, acc2 = {}, acc3 = {};

  v8bf* curp = (v8bf*)ldsB[0];
  v8bf* nxtp = (v8bf*)ldsB[1];

  // Prologue: stage slice 0.
  curp[sslot] = *(const v8bf*)(bsrc);
  __syncthreads();

  // ---- h part: K-slices 0..31 (k0 = 0..1023).
  for (int i = 0; i < 32; ++i) {
    const int k0 = i << 5;
    v8bf bnext = *(const v8bf*)(bsrc + k0 + 32);         // prefetch slice i+1
    v8bf alo = *(const v8bf*)(hrow + k0);
    v8bf ahi = *(const v8bf*)(hrow + k0 + 16);
    gate_mma(curp + (lane << 1), cat8(alo, ahi), acc0, acc1, acc2, acc3);
    nxtp[sslot] = bnext;
    __syncthreads();
    v8bf* tmp = curp; curp = nxtp; nxtp = tmp;
  }

  // ---- x part: K-slices 32..63 (k0 = 1024..2047). Final prefetch overreads
  //      16B past WcatT into the h0 buffer that follows it in d_ws (unused).
  for (int i = 0; i < 32; ++i) {
    const int k0 = 1024 + (i << 5);
    v8bf bnext = *(const v8bf*)(bsrc + k0 + 32);
    const float* xp = xrow + (i << 5);
    v4f f0 = *(const v4f*)(xp);
    v4f f1 = *(const v4f*)(xp + 4);
    v4f f2 = *(const v4f*)(xp + 16);
    v4f f3 = *(const v4f*)(xp + 20);
    v16bf afrag;
#pragma unroll
    for (int e = 0; e < 4; ++e) {
      afrag[e]      = (__bf16)f0[e];
      afrag[4 + e]  = (__bf16)f1[e];
      afrag[8 + e]  = (__bf16)f2[e];
      afrag[12 + e] = (__bf16)f3[e];
    }
    gate_mma(curp + (lane << 1), afrag, acc0, acc1, acc2, acc3);
    nxtp[sslot] = bnext;
    __syncthreads();
    v8bf* tmp = curp; curp = nxtp; nxtp = tmp;
  }

  // ---- Fused LSTM pointwise update. C/D layout: VGPR v -> row v + 8*half.
  const int col = cbase + l15;
  const float bi = bias[col];
  const float bf = bias[1024 + col];
  const float bo = bias[2048 + col];
  const float bu = bias[3072 + col];

  float hvals[8];
#pragma unroll
  for (int v = 0; v < 8; ++v) {
    const int row = (wave << 4) + (half << 3) + v;
    const int idx = (row << 10) + col;
    float gi = fast_sigmoid(acc0[v] + bi);
    float gf = fast_sigmoid(acc1[v] + bf);
    float go = fast_sigmoid(acc2[v] + bo);
    float gu = fast_tanh   (acc3[v] + bu);
    float cn = gf * c_state[idx] + gi * gu;
    c_state[idx] = cn;
    float hn = go * fast_tanh(cn);
    hvals[v] = hn;
    h_out[idx] = (__bf16)hn;             // bf16 h for next step's A-matrix
  }
  if (write_out) {                       // uniform: only the final step
#pragma unroll
    for (int v = 0; v < 8; ++v) {
      const int row = (wave << 4) + (half << 3) + v;
      h_f32[(row << 10) + col] = hvals[v];
    }
  }
}

extern "C" void kernel_launch(void* const* d_in, const int* in_sizes, int n_in,
                              void* d_out, int out_size, void* d_ws, size_t ws_size,
                              hipStream_t stream) {
  (void)in_sizes; (void)n_in; (void)out_size; (void)ws_size;
  const float* x    = (const float*)d_in[0];   // [128,512,1024]
  // d_in[1] = adjacency, unused by the reference forward
  const float* Wx   = (const float*)d_in[2];   // [4,1024,1024]
  const float* Wh   = (const float*)d_in[3];   // [4,1024,1024]
  const float* bias = (const float*)d_in[4];   // [4,1024]
  float* out = (float*)d_out;                  // [128,1024]

  // Workspace layout (~17.8 MB total):
  __bf16* wcat = (__bf16*)d_ws;                         // 4096*2048 bf16 = 16 MiB
  __bf16* h0   = wcat + (size_t)GCOLS * KTOT;           // 128*1024 bf16
  __bf16* h1   = h0 + BATCH * DIM;                      // 128*1024 bf16
  float*  c    = (float*)(h1 + BATCH * DIM);            // 128*1024 f32

  prep_weights<<<(GCOLS * KTOT) / 256, 256, 0, stream>>>(Wx, Wh, wcat);
  init_state<<<(BATCH * DIM) / 256, 256, 0, stream>>>(h0, c);

  for (int t = 0; t < TSEQ; ++t) {
    const __bf16* hin = (t & 1) ? h1 : h0;
    __bf16*      hout = (t & 1) ? h0 : h1;
    lstm_step<<<DIM / 16, 256, 0, stream>>>(wcat, x, bias, hin, hout, c, out,
                                            t, (t == TSEQ - 1) ? 1 : 0);
  }
}
